// RecurrentSBP_23373212025330
// MI455X (gfx1250) — matrix operand order repaired and checked by reading, not compiled, not run on hardware
//
#include <hip/hip_runtime.h>
#include <hip/hip_bf16.h>
#include <math.h>

// ---------------------------------------------------------------------------
// RecurrentSBP for MI455X (gfx1250): bf16 WMMA GEMMs, software-pipelined
// global->reg->LDS(bf16) staging with fence-free split barriers so global
// loads stay in flight across the WMMA section; async-LDS decoder.
// ---------------------------------------------------------------------------

typedef __attribute__((ext_vector_type(16))) __bf16 v16bf;
typedef __attribute__((ext_vector_type(8)))  __bf16 v8bf;
typedef __attribute__((ext_vector_type(4)))  __bf16 v4bf;
typedef __attribute__((ext_vector_type(8)))  float  v8f;

#define DIN_   49152
#define HDIM_  512
#define ZDIM_  64
#define B_     256
#define KSEG_  8
#define HW_    16384

// ---- CDNA5 async global->LDS copy (ASYNCcnt-tracked) ----------------------
__device__ __forceinline__ void async_copy_b128(void* lds_dst, const void* gsrc) {
  unsigned lds_off = (unsigned)(unsigned long long)lds_dst;  // low 32 bits = LDS addr
  asm volatile("global_load_async_to_lds_b128 %0, %1, off"
               :: "v"(lds_off), "v"(gsrc) : "memory");
}
__device__ __forceinline__ void wait_async0() {
  asm volatile("s_wait_asynccnt 0" ::: "memory");
}
// raw workgroup split barrier: NO implicit vmcnt/loadcnt fence, so global
// loads issued before it remain outstanding across the barrier.
__device__ __forceinline__ void wg_barrier() {
  asm volatile("s_barrier_signal -1\n\ts_barrier_wait -1" ::: "memory");
}
__device__ __forceinline__ void wait_ds0() {
  asm volatile("s_wait_dscnt 0" ::: "memory");
}

__device__ __forceinline__ v16bf combine16(v8bf lo, v8bf hi) {
  return __builtin_shufflevector(lo, hi, 0, 1, 2, 3, 4, 5, 6, 7,
                                 8, 9, 10, 11, 12, 13, 14, 15);
}

__device__ __forceinline__ float sigm(float x)      { return 1.f / (1.f + expf(-x)); }
__device__ __forceinline__ float softplusf(float x) { return (x > 20.f) ? x : log1pf(expf(x)); }
__device__ __forceinline__ float logsigf(float x)   { return fminf(x, 0.f) - log1pf(expf(-fabsf(x))); }

// ---------------------------------------------------------------------------
// Generic bf16-WMMA GEMM: C[M,N] = act(A @ B + bias); A fp32 (optional concat
// A0[:, :K1] | A1[:, K1:]), B fp32 row-major [K,N].
// Block tile 128(M) x 128(N), K-chunk 32; 256 thr = 8 waves; wave w owns rows
// [16w,16w+16) x 128 cols (8 WMMA n-tiles).
// Software pipeline per chunk:
//   barrier (consumers done) -> cvt regs -> LDS -> issue NEXT global loads
//   -> s_wait_dscnt 0 -> barrier (LDS ready, NO load fence) -> 8 WMMAs
// so the next chunk's global loads overlap the WMMA section.
// grid = (M/128, N/128, ksplits); atomic mode accumulates fp32 partials.
// ---------------------------------------------------------------------------
__global__ __launch_bounds__(256)
void gemm_wmma_kernel(const float* __restrict__ A0, const float* __restrict__ A1,
                      int K1, int lda0, int lda1,
                      const float* __restrict__ Bm, int ldb,
                      const float* __restrict__ bias,
                      float* __restrict__ C, int ldc,
                      int klen, int act, int do_atomic)
{
  __shared__ __bf16 As[128 * 32];   // 8 KB, row-major [m][k]
  __shared__ __bf16 Bt[128 * 32];   // 8 KB, transposed [n][k]

  const int tid  = threadIdx.x;
  const int wave = tid >> 5;
  const int lane = tid & 31;
  const int row0 = blockIdx.x * 128;
  const int col0 = blockIdx.y * 128;
  const int kbase = blockIdx.z * klen;
  const int kend  = kbase + klen;

  v8f acc[8];
#pragma unroll
  for (int t = 0; t < 8; ++t)
#pragma unroll
    for (int j = 0; j < 8; ++j) acc[t][j] = 0.f;

  // ---- per-thread staging coordinates (fixed for the whole kernel) ----
  const int ra  = tid >> 3;          // A row (i-th vec adds 32)
  const int c4a = (tid & 7) << 2;    // A col within chunk
  const int rb  = tid >> 5;          // B k-row (i-th vec adds 8)
  const int cb  = (tid & 31) << 2;   // B col within tile

  // both concat sides precomputed; advanced by +32 floats per chunk.
  // ap1 may point below A1 until kc reaches K1 (never dereferenced before).
  const float* ap0 = A0 + (ptrdiff_t)(row0 + ra) * lda0 + kbase + c4a;
  const float* ap1 = A1 + (ptrdiff_t)(row0 + ra) * lda1 + (ptrdiff_t)(kbase - K1) + c4a;
  const float* bp  = Bm + (ptrdiff_t)(kbase + rb) * ldb + col0 + cb;

  const int mrow = wave * 16 + (lane & 15);
  const int kbA  = (lane < 16) ? 0 : 8;    // ISA 16-bit A 16x32 striping
  const int kbB  = (lane < 16) ? 0 : 8;    // ISA 16-bit B 32x16 striping
  const int ncol = lane & 15;

  float4 ga[4], gb[4];
  // ---- prime the pipeline: load chunk 0 into registers ----
  {
    const float* ap = (kbase < K1) ? ap0 : ap1;
    const int    la = (kbase < K1) ? lda0 : lda1;
#pragma unroll
    for (int i = 0; i < 4; ++i)
      ga[i] = *(const float4*)(ap + (ptrdiff_t)(32 * i) * la);
#pragma unroll
    for (int i = 0; i < 4; ++i)
      gb[i] = *(const float4*)(bp + (ptrdiff_t)(8 * i) * ldb);
  }

  for (int kc = kbase; kc < kend; kc += 32) {
    wg_barrier();      // previous chunk's consumers done with LDS (no fence)

    // ---- convert staged registers -> bf16 LDS ----
#pragma unroll
    for (int i = 0; i < 4; ++i) {
      v4bf h;
      h[0] = (__bf16)ga[i].x; h[1] = (__bf16)ga[i].y;
      h[2] = (__bf16)ga[i].z; h[3] = (__bf16)ga[i].w;
      *(v4bf*)&As[(ra + 32 * i) * 32 + c4a] = h;
    }
#pragma unroll
    for (int i = 0; i < 4; ++i) {
      int r = rb + 8 * i;
      Bt[(cb + 0) * 32 + r] = (__bf16)gb[i].x;
      Bt[(cb + 1) * 32 + r] = (__bf16)gb[i].y;
      Bt[(cb + 2) * 32 + r] = (__bf16)gb[i].z;
      Bt[(cb + 3) * 32 + r] = (__bf16)gb[i].w;
    }

    // ---- issue next chunk's global loads (remain in flight across the
    //      barrier below and the WMMA section) ----
    ap0 += 32; ap1 += 32; bp += (ptrdiff_t)32 * ldb;
    if (kc + 32 < kend) {
      const float* ap = (kc + 32 < K1) ? ap0 : ap1;
      const int    la = (kc + 32 < K1) ? lda0 : lda1;
#pragma unroll
      for (int i = 0; i < 4; ++i)
        ga[i] = *(const float4*)(ap + (ptrdiff_t)(32 * i) * la);
#pragma unroll
      for (int i = 0; i < 4; ++i)
        gb[i] = *(const float4*)(bp + (ptrdiff_t)(8 * i) * ldb);
    }

    wait_ds0();        // own LDS stores committed
    wg_barrier();      // LDS tiles visible to all waves (loads still in flight)

    // ---- A fragment: two contiguous 16B vector loads ----
    v16bf afrag = combine16(*(const v8bf*)&As[mrow * 32 + kbA],
                            *(const v8bf*)&As[mrow * 32 + 16 + kbA]);
    // ---- 8 n-tiles, A-reuse hint on all but the last ----
    {
      v16bf bfrag = combine16(*(const v8bf*)&Bt[(0 * 16 + ncol) * 32 + kbB],
                              *(const v8bf*)&Bt[(0 * 16 + ncol) * 32 + 16 + kbB]);
      acc[0] = __builtin_amdgcn_wmma_f32_16x16x32_bf16(
          false, afrag, false, bfrag, (short)0, acc[0], true, false);
    }
#pragma unroll
    for (int t = 1; t < 7; ++t) {
      v16bf bfrag = combine16(*(const v8bf*)&Bt[(t * 16 + ncol) * 32 + kbB],
                              *(const v8bf*)&Bt[(t * 16 + ncol) * 32 + 16 + kbB]);
      acc[t] = __builtin_amdgcn_wmma_f32_16x16x32_bf16(
          false, afrag, false, bfrag, (short)0, acc[t], true, false);
    }
    {
      v16bf bfrag = combine16(*(const v8bf*)&Bt[(7 * 16 + ncol) * 32 + kbB],
                              *(const v8bf*)&Bt[(7 * 16 + ncol) * 32 + 16 + kbB]);
      acc[7] = __builtin_amdgcn_wmma_f32_16x16x32_bf16(
          false, afrag, false, bfrag, (short)0, acc[7], false, false);
    }
  }

  // ---- epilogue (C layout: VGPR j -> M=j lanes 0-15 / M=j+8 lanes 16-31) ----
  const int rbase = wave * 16 + ((lane < 16) ? 0 : 8);
#pragma unroll
  for (int t = 0; t < 8; ++t) {
    int col = col0 + t * 16 + ncol;
#pragma unroll
    for (int j = 0; j < 8; ++j) {
      int row = row0 + rbase + j;
      float v = acc[t][j];
      if (do_atomic) {
        atomicAdd(&C[(size_t)row * ldc + col], v);
      } else {
        if (bias) v += bias[col];
        if (act == 1) v = fmaxf(v, 0.f);
        C[(size_t)row * ldc + col] = v;
      }
    }
  }
}

// ---------------------------------------------------------------------------
// Fused decoder: zs[8,256,64] @ W_dec[64,16384] + b_dec, then log-sigmoid and
// the exclusive K-cumsum, writing log_ms[8,256,16384] directly (no 134 MB
// intermediate). Block = 128 thr (4 waves). M-rows 0..7 -> (k,b0), 8..15 ->
// (k,b0+1): the cumsum over k is per-lane register arithmetic over acc[0..7].
// f32 tiles arrive via CDNA5 async-LDS copies, then one LDS bf16 conversion
// pass (W transposed) so fragments are contiguous vector loads.
// ---------------------------------------------------------------------------
__global__ __launch_bounds__(128)
void decoder_kernel(const float* __restrict__ zs, const float* __restrict__ Wd,
                    const float* __restrict__ bd, float* __restrict__ out)
{
  __shared__ float  Zt[16 * 64];    //  4 KB f32 A tile (rows = (k,b) pairs)
  __shared__ float  Wt[64 * 128];   // 32 KB f32 B tile
  __shared__ __bf16 Zb[16 * 64];    //  2 KB bf16 row-major
  __shared__ __bf16 Wb[128 * 64];   // 16 KB bf16 transposed [n][k]

  const int tid  = threadIdx.x;
  const int wave = tid >> 5;
  const int lane = tid & 31;
  const int b0   = blockIdx.x * 2;
  const int p0   = blockIdx.y * 128;

  // async stage Z tile: row r -> k=r&7, b=b0+(r>>3)
#pragma unroll
  for (int i = 0; i < 2; ++i) {
    int v  = tid + 128 * i;            // 0..255 vec4s
    int r  = v >> 4;
    int c4 = (v & 15) << 2;
    int k  = r & 7, b = b0 + (r >> 3);
    async_copy_b128(&Zt[r * 64 + c4], zs + ((size_t)k * B_ + b) * ZDIM_ + c4);
  }
  // async stage W_dec tile 64x128
#pragma unroll
  for (int i = 0; i < 16; ++i) {
    int v  = tid + 128 * i;            // 0..2047 vec4s
    int r  = v >> 5;
    int c4 = (v & 31) << 2;
    async_copy_b128(&Wt[r * 128 + c4], Wd + (size_t)r * HW_ + p0 + c4);
  }
  wait_async0();
  wg_barrier();

  // f32 -> bf16 conversion pass
#pragma unroll
  for (int j = 0; j < 8; ++j) {
    int e = tid * 8 + j;               // 0..1023
    Zb[e] = (__bf16)Zt[e];
  }
#pragma unroll
  for (int k = 0; k < 64; ++k)         // col = tid, transpose W
    Wb[tid * 64 + k] = (__bf16)Wt[k * 128 + tid];
  wait_ds0();
  wg_barrier();

  const int mrow = lane & 15;
  const int kbA  = (lane < 16) ? 0 : 8;
  const int kbB  = (lane < 16) ? 0 : 8;
  const int ncol = lane & 15;

  v16bf a0 = combine16(*(const v8bf*)&Zb[mrow * 64 + kbA],
                       *(const v8bf*)&Zb[mrow * 64 + 16 + kbA]);
  v16bf a1 = combine16(*(const v8bf*)&Zb[mrow * 64 + 32 + kbA],
                       *(const v8bf*)&Zb[mrow * 64 + 48 + kbA]);

#pragma unroll
  for (int tt = 0; tt < 2; ++tt) {
    const int nt  = wave * 2 + tt;     // 8 n-tiles of 16 cols across 4 waves
    const int col = p0 + nt * 16 + ncol;
    v8f acc;
#pragma unroll
    for (int j = 0; j < 8; ++j) acc[j] = 0.f;

    v16bf b0f = combine16(*(const v8bf*)&Wb[(nt * 16 + ncol) * 64 + kbB],
                          *(const v8bf*)&Wb[(nt * 16 + ncol) * 64 + 16 + kbB]);
    v16bf b1f = combine16(*(const v8bf*)&Wb[(nt * 16 + ncol) * 64 + 32 + kbB],
                          *(const v8bf*)&Wb[(nt * 16 + ncol) * 64 + 48 + kbB]);
    acc = __builtin_amdgcn_wmma_f32_16x16x32_bf16(false, a0, false, b0f,
                                                  (short)0, acc, true, false);
    acc = __builtin_amdgcn_wmma_f32_16x16x32_bf16(false, a1, false, b1f,
                                                  (short)0, acc, false, false);

    // epilogue: acc[j] = dec for k=j at (b,col); K-cumsum in registers
    const float bias = bd[col];
    const int   b    = b0 + ((lane < 16) ? 0 : 1);
    float run = 0.f;
#pragma unroll
    for (int k = 0; k < 8; ++k) {
      float dec = acc[k] + bias;
      float ls  = logsigf(dec);
      float lns = logsigf(-dec);
      out[((size_t)k * B_ + b) * HW_ + col] = run + ls;   // excl-prefix + ls
      run += lns;
    }
  }
}

// ---------------------------------------------------------------------------
// Elementwise helpers
// ---------------------------------------------------------------------------
__global__ void zero_kernel(float* __restrict__ p, int n) {
  int i = blockIdx.x * blockDim.x + threadIdx.x;
  if (i < n) p[i] = 0.f;
}

__global__ void bias_relu_kernel(float* __restrict__ h, const float* __restrict__ b,
                                 int n, int ld) {
  int i = blockIdx.x * blockDim.x + threadIdx.x;
  if (i < n) h[i] = fmaxf(h[i] + b[i % ld], 0.f);
}

// gates g[256,512] (i,f,g,o each 128 wide; zero init state -> f unused)
__global__ void lstm_gate_kernel(const float* __restrict__ g, float* __restrict__ h) {
  int idx = blockIdx.x * blockDim.x + threadIdx.x;   // 0..32767
  int r = idx >> 7, c = idx & 127;
  float gi = g[r * 512 + c];
  float gg = g[r * 512 + 256 + c];
  float go = g[r * 512 + 384 + c];
  float cc = sigm(gi) * tanhf(gg);
  h[r * 128 + c] = sigm(go) * tanhf(cc);
}

__global__ void z0_kernel(const float* __restrict__ enc, const float* __restrict__ eps0,
                          float* __restrict__ z0, float* __restrict__ out_kl) {
  int idx = blockIdx.x * blockDim.x + threadIdx.x;   // 0..16383
  int b = idx >> 6, j = idx & 63;
  float mu = enc[b * 128 + j];
  float lv = enc[b * 128 + 64 + j];
  float s  = softplusf(lv + 0.5f) + 1e-8f;
  z0[b * ZDIM_ + j] = mu + s * eps0[b * ZDIM_ + j];
  out_kl[(size_t)(b * ZDIM_ + j) * KSEG_ + 0] =
      -logf(s) + (s * s + mu * mu) * 0.5f - 0.5f;
}

__global__ void post_kernel(const float* __restrict__ encq, const float* __restrict__ encp,
                            const float* __restrict__ eq, float* __restrict__ zq,
                            float* __restrict__ out_kl, int t) {
  int idx = blockIdx.x * blockDim.x + threadIdx.x;   // 0..16383
  int b = idx >> 6, j = idx & 63;
  float mq = encq[b * 128 + j], lq = encq[b * 128 + 64 + j];
  float mp = encp[b * 128 + j], lp = encp[b * 128 + 64 + j];
  float sq  = softplusf(lq + 0.5f) + 1e-8f;
  float sp_ = softplusf(lp + 0.5f) + 1e-8f;
  zq[b * ZDIM_ + j] = mq + sq * eq[b * ZDIM_ + j];
  float d = mq - mp;
  out_kl[(size_t)(b * ZDIM_ + j) * KSEG_ + t] =
      logf(sp_ / sq) + (sq * sq + d * d) / (2.f * sp_ * sp_) - 0.5f;
}

// ---------------------------------------------------------------------------
extern "C" void kernel_launch(void* const* d_in, const int* in_sizes, int n_in,
                              void* d_out, int out_size, void* d_ws, size_t ws_size,
                              hipStream_t stream) {
  (void)in_sizes; (void)n_in; (void)out_size; (void)ws_size;

  const float* x      = (const float*)d_in[0];
  const float* eps0   = (const float*)d_in[2];
  const float* eps_q  = (const float*)d_in[3];
  const float* W_stem = (const float*)d_in[5];
  const float* b_stem = (const float*)d_in[6];
  const float* W_g    = (const float*)d_in[7];
  const float* b_g    = (const float*)d_in[8];
  const float* Wih_q  = (const float*)d_in[9];
  const float* b_q    = (const float*)d_in[10];
  const float* Wlin_q = (const float*)d_in[11];
  const float* blin_q = (const float*)d_in[12];
  const float* Wih_p  = (const float*)d_in[13];
  const float* b_p    = (const float*)d_in[14];
  const float* Wlin_p = (const float*)d_in[15];
  const float* blin_p = (const float*)d_in[16];
  const float* W_dec  = (const float*)d_in[17];
  const float* b_dec  = (const float*)d_in[18];

  float* out    = (float*)d_out;
  float* out_kl = out + (size_t)KSEG_ * B_ * HW_;

  // workspace layout (fp32)
  float* ws   = (float*)d_ws;
  float* hf   = ws;              // 256*512
  float* enc  = hf   + 131072;   // 256*128
  float* zs   = enc  + 32768;    // 8*256*64
  float* gq   = zs   + 131072;   // 256*512
  float* hq   = gq   + 131072;   // 256*128
  float* encq = hq   + 32768;    // 256*128
  float* gp   = encq + 32768;    // 256*512
  float* hp   = gp   + 131072;   // 256*128
  float* encp = hp   + 32768;    // 256*128

  const int BIG = 1 << 30;       // "no concat" sentinel for K1

  // 1) stem: hf_acc = x @ W_stem  (split-K x16, fp32 atomic partials)
  zero_kernel<<<dim3(131072 / 256), 256, 0, stream>>>(hf, 131072);
  gemm_wmma_kernel<<<dim3(2, 4, 16), 256, 0, stream>>>(
      x, x, BIG, DIN_, DIN_, W_stem, HDIM_, nullptr, hf, HDIM_,
      DIN_ / 16, /*act*/0, /*atomic*/1);
  bias_relu_kernel<<<dim3(512), 256, 0, stream>>>(hf, b_stem, 131072, HDIM_);

  // 2) gaussian head: enc = hf @ W_g + b_g
  gemm_wmma_kernel<<<dim3(2, 1, 1), 256, 0, stream>>>(
      hf, hf, BIG, HDIM_, HDIM_, W_g, 128, b_g, enc, 128, HDIM_, 0, 0);
  z0_kernel<<<dim3(64), 256, 0, stream>>>(enc, eps0, zs, out_kl);

  // 3) sequential scan over K-1 segments
  for (int t = 1; t < KSEG_; ++t) {
    const float* zprev = zs + (size_t)(t - 1) * B_ * ZDIM_;
    float*       znew  = zs + (size_t)t * B_ * ZDIM_;
    // gates_q = [hf | zprev] @ Wih_q + b_q   (concat split K1=512)
    gemm_wmma_kernel<<<dim3(2, 4, 1), 256, 0, stream>>>(
        hf, zprev, HDIM_, HDIM_, ZDIM_, Wih_q, 512, b_q, gq, 512, 576, 0, 0);
    lstm_gate_kernel<<<dim3(128), 256, 0, stream>>>(gq, hq);
    gemm_wmma_kernel<<<dim3(2, 1, 1), 256, 0, stream>>>(
        hq, hq, BIG, 128, 128, Wlin_q, 128, blin_q, encq, 128, 128, 0, 0);
    // gates_p = zprev @ Wih_p + b_p
    gemm_wmma_kernel<<<dim3(2, 4, 1), 256, 0, stream>>>(
        zprev, zprev, BIG, ZDIM_, ZDIM_, Wih_p, 512, b_p, gp, 512, ZDIM_, 0, 0);
    lstm_gate_kernel<<<dim3(128), 256, 0, stream>>>(gp, hp);
    gemm_wmma_kernel<<<dim3(2, 1, 1), 256, 0, stream>>>(
        hp, hp, BIG, 128, 128, Wlin_p, 128, blin_p, encp, 128, 128, 0, 0);
    post_kernel<<<dim3(64), 256, 0, stream>>>(
        encq, encp, eps_q + (size_t)(t - 1) * B_ * ZDIM_, znew, out_kl, t);
  }

  // 4) fused decoder + log-sigmoid + exclusive K-cumsum -> log_ms
  decoder_kernel<<<dim3(B_ / 2, HW_ / 128), 128, 0, stream>>>(zs, W_dec, b_dec, out);
}
